// DGLMAG240RGNN_57492432224403
// MI455X (gfx1250) — matrix operand order repaired and checked by reading, not compiled, not run on hardware
//
#include <hip/hip_runtime.h>
#include <cstdint>
#include <cstddef>

// ---------------- types ----------------
typedef __bf16 bf16_t;
typedef __attribute__((ext_vector_type(16))) __bf16 v16bf;
typedef __attribute__((ext_vector_type(8)))  __bf16 v8bf;
typedef __attribute__((ext_vector_type(8)))  float  v8f;

#define N_SRC0 200000
#define N_DST0 20000
#define N_DST1 2048
#define E0_N   400000
#define E1_N   100000
#define IN_C   768
#define HID    1024
#define OUT_C  153
#define OUT_PAD 192            // padded to 12 N-tiles so every wave chunk is a full 4 tiles
#define N_ET   5
#define BN_EPS 1e-5f

__device__ __forceinline__ bf16_t f2bf(float f) {
    union { float f; unsigned u; } uf; uf.f = f;
    unsigned u = uf.u;
    unsigned r = (u + 0x7FFFu + ((u >> 16) & 1u)) >> 16;   // RNE
    union { unsigned short s; bf16_t b; } ub; ub.s = (unsigned short)r;
    return ub.b;
}
__device__ __forceinline__ float bf2f(bf16_t b) {
    union { unsigned short s; bf16_t b; } ub; ub.b = b;
    union { unsigned u; float f; } uf; uf.u = ((unsigned)ub.s) << 16;
    return uf.f;
}

// ---------------- utility kernels ----------------
__global__ void mag_zero_f32(float* __restrict__ p, size_t n) {
    for (size_t i = (size_t)blockIdx.x * blockDim.x + threadIdx.x; i < n;
         i += (size_t)gridDim.x * blockDim.x) p[i] = 0.f;
}

// bias_out[n] = skip_b[n] + sum_j b[j][n]
__global__ void mag_combine_bias(const float* __restrict__ skipB,
                                 const float* __restrict__ b, int N,
                                 float* __restrict__ out) {
    int i = blockIdx.x * blockDim.x + threadIdx.x;
    if (i < N) {
        float v = skipB[i];
        #pragma unroll
        for (int j = 0; j < N_ET; ++j) v += b[j * N + i];
        out[i] = v;
    }
}

__global__ void mag_pad_bias(const float* __restrict__ b, int N, int Npad,
                             float* __restrict__ out) {
    int i = blockIdx.x * blockDim.x + threadIdx.x;
    if (i < Npad) out[i] = (i < N) ? b[i] : 0.f;
}

// ---------------- B-operand fragment packing ----------------
// Packed layout: [kt][nt][lane][e] of bf16, lane = {khalf = lane>>4, col = lane&15},
// element e -> srcK = kt*32 + 16*khalf + e, srcN = nt*16 + col.
// SAGE B: rows [0, inC)           = skip_w + sum_j Wself[j]   (all x_dst GEMMs fused)
//         rows [inC*(1+j), +inC)  = Wneigh[j]
__global__ void mag_pack_sage_B(const float* __restrict__ skipW,
                                const float* __restrict__ Wself,
                                const float* __restrict__ Wneigh,
                                int inC, int N, int ntiles, int K,
                                bf16_t* __restrict__ out, size_t total) {
    for (size_t idx = (size_t)blockIdx.x * blockDim.x + threadIdx.x; idx < total;
         idx += (size_t)gridDim.x * blockDim.x) {
        int e    = (int)(idx & 15);
        int lane = (int)((idx >> 4) & 31);
        size_t tile = idx >> 9;
        int nt = (int)(tile % (size_t)ntiles);
        int kt = (int)(tile / (size_t)ntiles);
        int srcK = kt * 32 + ((lane >> 4) << 4) + e;
        int srcN = nt * 16 + (lane & 15);
        float val = 0.f;
        if (srcK < K && srcN < N) {
            if (srcK < inC) {
                val = skipW[(size_t)srcK * N + srcN];
                #pragma unroll
                for (int j = 0; j < N_ET; ++j)
                    val += Wself[((size_t)j * inC + srcK) * N + srcN];
            } else {
                int r = srcK - inC;
                int j = r / inC, kk = r % inC;
                val = Wneigh[((size_t)j * inC + kk) * N + srcN];
            }
        }
        out[idx] = f2bf(val);
    }
}

// Plain B: W is [K, Nlog] row-major; pad columns >= Nlog with zeros.
__global__ void mag_pack_plain_B(const float* __restrict__ W, int K, int Nlog,
                                 int ntiles, bf16_t* __restrict__ out, size_t total) {
    for (size_t idx = (size_t)blockIdx.x * blockDim.x + threadIdx.x; idx < total;
         idx += (size_t)gridDim.x * blockDim.x) {
        int e    = (int)(idx & 15);
        int lane = (int)((idx >> 4) & 31);
        size_t tile = idx >> 9;
        int nt = (int)(tile % (size_t)ntiles);
        int kt = (int)(tile / (size_t)ntiles);
        int srcK = kt * 32 + ((lane >> 4) << 4) + e;
        int srcN = nt * 16 + (lane & 15);
        float val = (srcK < K && srcN < Nlog) ? W[(size_t)srcK * Nlog + srcN] : 0.f;
        out[idx] = f2bf(val);
    }
}

// ---------------- A-panel producers ----------------
__global__ void mag_cast_to_A(const float* __restrict__ x, int M, int F,
                              bf16_t* __restrict__ A, int lda, int colOff) {
    size_t n = (size_t)M * F;
    for (size_t i = (size_t)blockIdx.x * blockDim.x + threadIdx.x; i < n;
         i += (size_t)gridDim.x * blockDim.x) {
        int r = (int)(i / (size_t)F), f = (int)(i % (size_t)F);
        A[(size_t)r * lda + colOff + f] = f2bf(x[i]);
    }
}

__global__ void mag_mean_to_A(const float* __restrict__ sums,
                              const float* __restrict__ cnts, int nd, int F,
                              bf16_t* __restrict__ A, int lda, int colOff) {
    size_t n = (size_t)nd * F;
    for (size_t i = (size_t)blockIdx.x * blockDim.x + threadIdx.x; i < n;
         i += (size_t)gridDim.x * blockDim.x) {
        int d = (int)(i / (size_t)F), f = (int)(i % (size_t)F);
        float c = cnts[d]; c = (c < 1.f) ? 1.f : c;
        A[(size_t)d * lda + colOff + f] = f2bf(sums[i] / c);
    }
}

// ---------------- edge aggregation (one etype per pass; accumulator L2-resident) ----------------
__global__ void mag_scatter_f32(const float* __restrict__ feats, int F,
                                const int* __restrict__ src, const int* __restrict__ dst,
                                const int* __restrict__ et, int E, int j,
                                float* __restrict__ sums, float* __restrict__ cnts) {
    int lane = threadIdx.x & 31;
    int wid  = blockIdx.x * (blockDim.x >> 5) + (threadIdx.x >> 5);
    int nw   = gridDim.x * (blockDim.x >> 5);
    for (int e = wid; e < E; e += nw) {
        if (et[e] != j) continue;
        int s = src[e], d = dst[e];
        const float* sp = feats + (size_t)s * F;
        float* dp = sums + (size_t)d * F;
        for (int f = lane; f < F; f += 32) atomicAdd(dp + f, sp[f]);
        if (lane == 0) atomicAdd(cnts + d, 1.0f);
    }
}

__global__ void mag_scatter_bf16(const bf16_t* __restrict__ feats, int F,
                                 const int* __restrict__ src, const int* __restrict__ dst,
                                 const int* __restrict__ et, int E, int j,
                                 float* __restrict__ sums, float* __restrict__ cnts) {
    int lane = threadIdx.x & 31;
    int wid  = blockIdx.x * (blockDim.x >> 5) + (threadIdx.x >> 5);
    int nw   = gridDim.x * (blockDim.x >> 5);
    for (int e = wid; e < E; e += nw) {
        if (et[e] != j) continue;
        int s = src[e], d = dst[e];
        const bf16_t* sp = feats + (size_t)s * F;
        float* dp = sums + (size_t)d * F;
        for (int f = lane; f < F; f += 32) atomicAdd(dp + f, bf2f(sp[f]));
        if (lane == 0) atomicAdd(cnts + d, 1.0f);
    }
}

// ---------------- bf16 WMMA GEMM ----------------
// C[M,Nlog] = A[M,K](bf16,row-major,lda) @ Bpacked + bias.
// block = 256 threads = 8 waves; block owns 16 rows; each wave chunk = exactly 4 N-tiles
// (NTILES is always a multiple of 4). No lane-divergent control flow around WMMA.
template <int NTILES>
__global__ void mag_wmma_gemm(const bf16_t* __restrict__ A, int lda,
                              const bf16_t* __restrict__ Bp,
                              const float* __restrict__ bias,
                              float* __restrict__ C, int ldc,
                              int Nlog, int K) {
    const int lane = threadIdx.x & 31;
    // wave id forced scalar: all control flow around WMMA is wave-uniform (EXEC all-1s)
    const int wave = __builtin_amdgcn_readfirstlane(threadIdx.x >> 5);
    const int m0   = blockIdx.x * 16;
    const int half = lane >> 4;
    // A 16x32 layout: lane holds row (lane&15); its 16 elements are the
    // contiguous k range k0 + 16*half .. k0 + 16*half + 15 (matches packed B).
    const bf16_t* arowBase = A + (size_t)(m0 + (lane & 15)) * lda + 16 * half;
    const int ktiles = K >> 5;

    for (int ntBase = wave * 4; ntBase < NTILES; ntBase += 32) {
        v8f acc0 = {}, acc1 = {}, acc2 = {}, acc3 = {};
        const bf16_t* aptr = arowBase;
        const bf16_t* bptr = Bp + ((size_t)ntBase * 32 + lane) * 16;

        for (int kt = 0; kt < ktiles; ++kt) {
            v8bf alo = *(const v8bf*)(aptr);
            v8bf ahi = *(const v8bf*)(aptr + 8);
            v16bf afrag = __builtin_shufflevector(alo, ahi,
                0, 1, 2, 3, 4, 5, 6, 7, 8, 9, 10, 11, 12, 13, 14, 15);

            v16bf b0 = *(const v16bf*)(bptr);
            v16bf b1 = *(const v16bf*)(bptr + 512);
            v16bf b2 = *(const v16bf*)(bptr + 1024);
            v16bf b3 = *(const v16bf*)(bptr + 1536);
            __builtin_prefetch(bptr + (size_t)NTILES * 512, 0, 1);  // next K-tile of B

            acc0 = __builtin_amdgcn_wmma_f32_16x16x32_bf16(
                false, afrag, false, b0, (short)0, acc0, false, false);
            acc1 = __builtin_amdgcn_wmma_f32_16x16x32_bf16(
                false, afrag, false, b1, (short)0, acc1, false, false);
            acc2 = __builtin_amdgcn_wmma_f32_16x16x32_bf16(
                false, afrag, false, b2, (short)0, acc2, false, false);
            acc3 = __builtin_amdgcn_wmma_f32_16x16x32_bf16(
                false, afrag, false, b3, (short)0, acc3, false, false);

            aptr += 32;
            bptr += (size_t)NTILES * 512;
        }

        // C layout: element v -> row m0 + v + 8*half, col = nt*16 + (lane&15)
        auto storeTile = [&](v8f a, int t) {
            int col = (ntBase + t) * 16 + (lane & 15);
            if (col < Nlog) {
                float bv = bias[col];
                #pragma unroll
                for (int v = 0; v < 8; ++v)
                    C[(size_t)(m0 + v + 8 * half) * ldc + col] = a[v] + bv;
            }
        };
        storeTile(acc0, 0);
        storeTile(acc1, 1);
        storeTile(acc2, 2);
        storeTile(acc3, 3);
    }
}

// ---------------- batchnorm (batch stats, biased variance) ----------------
__global__ void mag_bn_stats(const float* __restrict__ h, int M, int Ncol,
                             int rowsPerBlock, float* __restrict__ s,
                             float* __restrict__ s2) {
    int col = blockIdx.x * blockDim.x + threadIdx.x;
    if (col >= Ncol) return;
    int r0 = blockIdx.y * rowsPerBlock;
    int r1 = r0 + rowsPerBlock; if (r1 > M) r1 = M;
    float a = 0.f, b = 0.f;
    for (int r = r0; r < r1; ++r) {
        float v = h[(size_t)r * Ncol + col];
        a += v; b += v * v;
    }
    atomicAdd(&s[col], a);
    atomicAdd(&s2[col], b);
}

// normalize -> affine -> ELU/ReLU -> bf16 ; optionally mirror first M2 rows into A-panel (stride lda2)
__global__ void mag_bn_act_apply(const float* __restrict__ h, int M, int Ncol,
                                 const float* __restrict__ s, const float* __restrict__ s2,
                                 const float* __restrict__ g, const float* __restrict__ b,
                                 int relu, bf16_t* __restrict__ out_bf,
                                 bf16_t* __restrict__ out2, int lda2, int M2) {
    size_t n = (size_t)M * Ncol;
    float invM = 1.f / (float)M;
    for (size_t i = (size_t)blockIdx.x * blockDim.x + threadIdx.x; i < n;
         i += (size_t)gridDim.x * blockDim.x) {
        int r = (int)(i / (size_t)Ncol), c = (int)(i % (size_t)Ncol);
        float mu  = s[c] * invM;
        float var = s2[c] * invM - mu * mu;
        float v = (h[i] - mu) * rsqrtf(var + BN_EPS) * g[c] + b[c];
        v = relu ? fmaxf(v, 0.f) : (v > 0.f ? v : (__expf(v) - 1.f));
        bf16_t bv = f2bf(v);
        out_bf[i] = bv;
        if (out2 != nullptr && r < M2) out2[(size_t)r * lda2 + c] = bv;
    }
}

// ---------------- host orchestration ----------------
static inline size_t alignUp256(size_t x) { return (x + 255) & ~(size_t)255; }
static inline int gblocks(size_t n) {
    size_t b = (n + 255) / 256;
    return (int)(b < 1 ? 1 : b);
}

extern "C" void kernel_launch(void* const* d_in, const int* in_sizes, int n_in,
                              void* d_out, int out_size, void* d_ws, size_t ws_size,
                              hipStream_t stream) {
    (void)in_sizes; (void)n_in; (void)out_size; (void)ws_size;
    // inputs (setup_inputs dict order)
    const float* x       = (const float*)d_in[0];
    const int*   src0    = (const int*)d_in[1];
    const int*   dst0    = (const int*)d_in[2];
    const int*   et0     = (const int*)d_in[3];
    const int*   src1    = (const int*)d_in[4];
    const int*   dst1    = (const int*)d_in[5];
    const int*   et1     = (const int*)d_in[6];
    const float* Wself0  = (const float*)d_in[7];
    const float* Wneigh0 = (const float*)d_in[8];
    const float* b0      = (const float*)d_in[9];
    const float* skip_w0 = (const float*)d_in[10];
    const float* skip_b0 = (const float*)d_in[11];
    const float* g0      = (const float*)d_in[12];
    const float* be0     = (const float*)d_in[13];
    const float* Wself1  = (const float*)d_in[14];
    const float* Wneigh1 = (const float*)d_in[15];
    const float* b1      = (const float*)d_in[16];
    const float* skip_w1 = (const float*)d_in[17];
    const float* skip_b1 = (const float*)d_in[18];
    const float* g1      = (const float*)d_in[19];
    const float* be1     = (const float*)d_in[20];
    const float* mlp_w1  = (const float*)d_in[21];
    const float* mlp_b1  = (const float*)d_in[22];
    const float* mlp_g   = (const float*)d_in[23];
    const float* mlp_be  = (const float*)d_in[24];
    const float* mlp_w2  = (const float*)d_in[25];
    const float* mlp_b2  = (const float*)d_in[26];
    float* out = (float*)d_out;

    // workspace carve-out
    char* w = (char*)d_ws;
    auto take = [&](size_t bytes) { char* p = w; w += alignUp256(bytes); return p; };

    const int K0 = IN_C * (1 + N_ET);   // 4608
    const int K1 = HID  * (1 + N_ET);   // 6144
    const int NT_HID = HID / 16;        // 64
    const int NT_OUT = OUT_PAD / 16;    // 12

    bf16_t* A0    = (bf16_t*)take((size_t)N_DST0 * K0 * 2);
    bf16_t* B0p   = (bf16_t*)take((size_t)K0 * HID * 2);
    float*  bias0 = (float*)take(HID * 4);
    float*  sums  = (float*)take((size_t)N_DST0 * IN_C * 4);   // reused by layer1 (smaller)
    float*  cnts  = (float*)take(N_DST0 * 4);
    float*  h0raw = (float*)take((size_t)N_DST0 * HID * 4);
    bf16_t* h0bf  = (bf16_t*)take((size_t)N_DST0 * HID * 2);
    float*  bnS   = (float*)take(2 * HID * 4);
    float*  bnS2  = bnS + HID;
    bf16_t* A1    = (bf16_t*)take((size_t)N_DST1 * K1 * 2);
    bf16_t* B1p   = (bf16_t*)take((size_t)K1 * HID * 2);
    float*  bias1 = (float*)take(HID * 4);
    float*  h1raw = (float*)take((size_t)N_DST1 * HID * 4);
    bf16_t* h1bf  = (bf16_t*)take((size_t)N_DST1 * HID * 2);
    bf16_t* B2p   = (bf16_t*)take((size_t)HID * HID * 2);
    float*  z1raw = (float*)take((size_t)N_DST1 * HID * 4);
    bf16_t* zbf   = (bf16_t*)take((size_t)N_DST1 * HID * 2);
    bf16_t* B3p   = (bf16_t*)take((size_t)HID * OUT_PAD * 2);
    float*  bias3 = (float*)take(OUT_PAD * 4);

    // ===== layer 0 =====
    {
        size_t tot = (size_t)K0 * HID;
        mag_pack_sage_B<<<gblocks(tot), 256, 0, stream>>>(skip_w0, Wself0, Wneigh0,
                                                          IN_C, HID, NT_HID, K0, B0p, tot);
    }
    mag_combine_bias<<<gblocks(HID), 256, 0, stream>>>(skip_b0, b0, HID, bias0);
    mag_cast_to_A<<<gblocks((size_t)N_DST0 * IN_C), 256, 0, stream>>>(
        x, N_DST0, IN_C, A0, K0, 0);
    for (int j = 0; j < N_ET; ++j) {
        mag_zero_f32<<<gblocks((size_t)N_DST0 * IN_C), 256, 0, stream>>>(
            sums, (size_t)N_DST0 * IN_C);
        mag_zero_f32<<<gblocks(N_DST0), 256, 0, stream>>>(cnts, (size_t)N_DST0);
        mag_scatter_f32<<<2048, 256, 0, stream>>>(x, IN_C, src0, dst0, et0, E0_N, j,
                                                  sums, cnts);
        mag_mean_to_A<<<gblocks((size_t)N_DST0 * IN_C), 256, 0, stream>>>(
            sums, cnts, N_DST0, IN_C, A0, K0, IN_C * (1 + j));
    }
    mag_wmma_gemm<64><<<N_DST0 / 16, 256, 0, stream>>>(A0, K0, B0p, bias0, h0raw, HID,
                                                       HID, K0);
    mag_zero_f32<<<gblocks(2 * HID), 256, 0, stream>>>(bnS, (size_t)2 * HID);
    mag_bn_stats<<<dim3(HID / 256, 40), 256, 0, stream>>>(h0raw, N_DST0, HID, 500,
                                                          bnS, bnS2);
    mag_bn_act_apply<<<gblocks((size_t)N_DST0 * HID), 256, 0, stream>>>(
        h0raw, N_DST0, HID, bnS, bnS2, g0, be0, /*relu=*/0,
        h0bf, A1, K1, N_DST1);

    // ===== layer 1 =====
    {
        size_t tot = (size_t)K1 * HID;
        mag_pack_sage_B<<<gblocks(tot), 256, 0, stream>>>(skip_w1, Wself1, Wneigh1,
                                                          HID, HID, NT_HID, K1, B1p, tot);
    }
    mag_combine_bias<<<gblocks(HID), 256, 0, stream>>>(skip_b1, b1, HID, bias1);
    for (int j = 0; j < N_ET; ++j) {
        mag_zero_f32<<<gblocks((size_t)N_DST1 * HID), 256, 0, stream>>>(
            sums, (size_t)N_DST1 * HID);
        mag_zero_f32<<<gblocks(N_DST1), 256, 0, stream>>>(cnts, (size_t)N_DST1);
        mag_scatter_bf16<<<1024, 256, 0, stream>>>(h0bf, HID, src1, dst1, et1, E1_N, j,
                                                   sums, cnts);
        mag_mean_to_A<<<gblocks((size_t)N_DST1 * HID), 256, 0, stream>>>(
            sums, cnts, N_DST1, HID, A1, K1, HID * (1 + j));
    }
    mag_wmma_gemm<64><<<N_DST1 / 16, 256, 0, stream>>>(A1, K1, B1p, bias1, h1raw, HID,
                                                       HID, K1);
    mag_zero_f32<<<gblocks(2 * HID), 256, 0, stream>>>(bnS, (size_t)2 * HID);
    mag_bn_stats<<<dim3(HID / 256, 8), 256, 0, stream>>>(h1raw, N_DST1, HID, 256,
                                                         bnS, bnS2);
    mag_bn_act_apply<<<gblocks((size_t)N_DST1 * HID), 256, 0, stream>>>(
        h1raw, N_DST1, HID, bnS, bnS2, g1, be1, /*relu=*/0,
        h1bf, (bf16_t*)nullptr, 0, 0);

    // ===== MLP head =====
    {
        size_t tot = (size_t)HID * HID;
        mag_pack_plain_B<<<gblocks(tot), 256, 0, stream>>>(mlp_w1, HID, HID, NT_HID,
                                                           B2p, tot);
    }
    mag_wmma_gemm<64><<<N_DST1 / 16, 256, 0, stream>>>(h1bf, HID, B2p, mlp_b1, z1raw,
                                                       HID, HID, HID);
    mag_zero_f32<<<gblocks(2 * HID), 256, 0, stream>>>(bnS, (size_t)2 * HID);
    mag_bn_stats<<<dim3(HID / 256, 8), 256, 0, stream>>>(z1raw, N_DST1, HID, 256,
                                                         bnS, bnS2);
    mag_bn_act_apply<<<gblocks((size_t)N_DST1 * HID), 256, 0, stream>>>(
        z1raw, N_DST1, HID, bnS, bnS2, mlp_g, mlp_be, /*relu=*/1,
        zbf, (bf16_t*)nullptr, 0, 0);

    {
        size_t tot = (size_t)HID * OUT_PAD;
        mag_pack_plain_B<<<gblocks(tot), 256, 0, stream>>>(mlp_w2, HID, OUT_C, NT_OUT,
                                                           B3p, tot);
    }
    mag_pad_bias<<<gblocks(OUT_PAD), 256, 0, stream>>>(mlp_b2, OUT_C, OUT_PAD, bias3);
    mag_wmma_gemm<12><<<N_DST1 / 16, 256, 0, stream>>>(zbf, HID, B3p, bias3, out,
                                                       OUT_C, OUT_C, HID);
}
